// LSHSelfAttention_8366596293018
// MI455X (gfx1250) — matrix operand order repaired and checked by reading, not compiled, not run on hardware
//
#include <hip/hip_runtime.h>
#include <hip/hip_bf16.h>
#include <math.h>
#include <stdint.h>

// ---------------- problem constants ----------------
#define T_LEN    8192
#define DIMV     512
#define NHEAD    8
#define DHEAD    64
#define NHASH_C  8
#define NBUCKETS 128      // buckets per hash round (2 * rotations)
#define BUCKET_SZ 64
#define NCHUNK   1024     // NHASH_C * NBUCKETS
#define FLAT_N   65536    // NHASH_C * T_LEN
#define NBLK     256      // sort blocks per bh (FLAT_N / 256)

typedef __attribute__((ext_vector_type(16))) __bf16 v16bf;
typedef __attribute__((ext_vector_type(8)))  float  v8f;

#if __has_builtin(__builtin_amdgcn_tensor_load_to_lds) && \
    __has_builtin(__builtin_amdgcn_s_wait_tensorcnt)
#define USE_TDM 1
#else
#define USE_TDM 0
#endif

#if USE_TDM
// =====================================================================
// TDM: program a 2-D tile descriptor (D#) and fire the Tensor Data
// Mover: rows x cols f32 tile at gsrc (row stride = row_stride_elems)
// -> LDS at lds_off, with 1 dword of LDS padding every 32 dwords
// (pad_interval=4 -> 32 dwords, pad_amount=0 -> 1 dword), i.e. LDS row
// stride = cols+1 floats. Tracked by TENSORcnt.
// 6-arg builtin form (this toolchain): (v4u, v8i, v4i, v4i, v8i, i32)
// =====================================================================
__device__ __forceinline__ void tdm_load_tile_f32(
    unsigned lds_off, const float* gsrc, int rows, int cols, int row_stride_elems)
{
    typedef __attribute__((ext_vector_type(4))) unsigned int v4u_t;
    typedef __attribute__((ext_vector_type(8))) int v8i_t;
    typedef __attribute__((ext_vector_type(4))) int v4i_t;
    unsigned long long ga = (unsigned long long)(uintptr_t)gsrc;
    v4u_t g0;
    g0[0] = 1u;                                            // count=1, user D#
    g0[1] = lds_off;                                       // LDS byte address
    g0[2] = (unsigned)(ga & 0xffffffffull);                // global_addr lo
    g0[3] = (unsigned)((ga >> 32) & 0x1ffffffull)          // global_addr hi
            | (2u << 30);                                  // type=2 ("image")
    v8i_t g1;
    g1[0] = (int)((2u << 16)                               // data_size = 4 B
                  | (1u << 20)                             // pad_enable
                  | (4u << 22));                           // pad_interval = 32 dw
    g1[1] = (int)(((unsigned)cols & 0xffffu) << 16);       // tensor_dim0 lo16
    g1[2] = (int)((((unsigned)cols >> 16) & 0xffffu)
                  | (((unsigned)rows & 0xffffu) << 16));   // dim0 hi | dim1 lo
    g1[3] = (int)((((unsigned)rows >> 16) & 0xffffu)
                  | (((unsigned)cols & 0xffffu) << 16));   // dim1 hi | tile_dim0
    g1[4] = (int)((unsigned)rows & 0xffffu);               // tile_dim1 (dim2=0)
    g1[5] = row_stride_elems;                              // dim0_stride lo32
    g1[6] = 0;
    g1[7] = 0;
    v4i_t g2 = {0, 0, 0, 0};
    v4i_t g3 = {0, 0, 0, 0};
    v8i_t g4 = {0, 0, 0, 0, 0, 0, 0, 0};
    __builtin_amdgcn_tensor_load_to_lds(g0, g1, g2, g3, g4, 0);
}
#endif

// =====================================================================
// Generic WMMA GEMM:  C[m,n] = sum_k A[m,k] * W[n,k]   (A,W f32 -> bf16)
// TDM double-buffered f32 tiles in LDS (K-step 32), WMMA bf16 compute.
// mode 0: C row-major [M][N]
// mode 1: C head-split [n/64][m][n%64]  (projection -> [head][t][dh])
// =====================================================================
__global__ __launch_bounds__(256) void gemm_bf16_kernel(
    const float* __restrict__ A, const float* __restrict__ W,
    float* __restrict__ C, int M, int N, int K, int mode)
{
    __shared__ float AsF[2][64][33];     // 33-float stride from TDM padding
    __shared__ float WsF[2][64][33];
    const int tid  = threadIdx.x;
    const int lane = tid & 31, wave = tid >> 5;
    const int l = lane & 15, hi = (lane >> 4) & 1;
    const int m0 = blockIdx.x * 64, n0 = blockIdx.y * 64;
    const float* Abase = A + (size_t)m0 * K;
    const float* Wbase = W + (size_t)n0 * K;
    const int nK = K >> 5;

    v8f acc[2];
#pragma unroll
    for (int s = 0; s < 2; ++s)
#pragma unroll
        for (int r = 0; r < 8; ++r) acc[s][r] = 0.0f;

#if USE_TDM
    if (wave == 0) {
        tdm_load_tile_f32((unsigned)(uintptr_t)&AsF[0][0][0], Abase, 64, 32, K);
        tdm_load_tile_f32((unsigned)(uintptr_t)&WsF[0][0][0], Wbase, 64, 32, K);
    }
#endif
    for (int kt = 0; kt < nK; ++kt) {
        const int cur = kt & 1;
#if USE_TDM
        if (wave == 0) {
            if (kt + 1 < nK) {          // issue next tile, then drain current
                tdm_load_tile_f32((unsigned)(uintptr_t)&AsF[cur ^ 1][0][0],
                                  Abase + (size_t)(kt + 1) * 32, 64, 32, K);
                tdm_load_tile_f32((unsigned)(uintptr_t)&WsF[cur ^ 1][0][0],
                                  Wbase + (size_t)(kt + 1) * 32, 64, 32, K);
                __builtin_amdgcn_s_wait_tensorcnt(2);
            } else {
                __builtin_amdgcn_s_wait_tensorcnt(0);
            }
        }
#else
        if (kt) __syncthreads();
#pragma unroll
        for (int i = 0; i < 8; ++i) {
            int e = tid + 256 * i;       // 2048 = 64*32 per matrix
            int r = e >> 5, cc = e & 31;
            AsF[cur][r][cc] = Abase[(size_t)r * K + kt * 32 + cc];
            WsF[cur][r][cc] = Wbase[(size_t)r * K + kt * 32 + cc];
        }
#endif
        __syncthreads();
#pragma unroll
        for (int s = 0; s < 2; ++s) {
            int tt = wave * 2 + s;
            int tm = tt >> 2, tn = tt & 3;
            v16bf a, b;
#pragma unroll
            for (int j = 0; j < 8; ++j) {
                int kb = hi * 16 + 2 * j;
                a[2 * j]     = (__bf16)AsF[cur][tm * 16 + l][kb];
                a[2 * j + 1] = (__bf16)AsF[cur][tm * 16 + l][kb + 1];
                int kc = hi * 16 + 2 * j;
                b[2 * j]     = (__bf16)WsF[cur][tn * 16 + l][kc];
                b[2 * j + 1] = (__bf16)WsF[cur][tn * 16 + l][kc + 1];
            }
            acc[s] = __builtin_amdgcn_wmma_f32_16x16x32_bf16(
                false, a, false, b, (short)0, acc[s], false, false);
        }
#if USE_TDM
        __syncthreads();                 // readers done before TDM rewrites
#endif
    }
#pragma unroll
    for (int s = 0; s < 2; ++s) {
        int tt = wave * 2 + s;
        int tm = tt >> 2, tn = tt & 3;
#pragma unroll
        for (int r = 0; r < 8; ++r) {
            int m = m0 + tm * 16 + r + hi * 8;
            int n = n0 + tn * 16 + l;
            float val = acc[s][r];
            if (mode == 0) {
                C[(size_t)m * N + n] = val;
            } else {
                int h = n >> 6;
                C[((size_t)h * T_LEN + m) * 64 + (n & 63)] = val;
            }
        }
    }
}

// =====================================================================
// LSH hash: bucket = argmax over concat(r, -r) of qk-row @ rotations
// keys[bh][r*T + t] = argmax + r*NBUCKETS   (in [0, 1024))
// =====================================================================
__global__ __launch_bounds__(256) void hash_kernel(
    const float* __restrict__ qk, const float* __restrict__ rot,
    int* __restrict__ keys)
{
    int gid = blockIdx.x * 256 + threadIdx.x;       // bh*NHASH*T
    int t  = gid & (T_LEN - 1);
    int r  = (gid >> 13) & 7;
    int bh = gid >> 16;
    const float* row = qk + ((size_t)bh * T_LEN + t) * 64;
    float q[64];
#pragma unroll
    for (int d = 0; d < 64; ++d) q[d] = row[d];
    float bp = -INFINITY, bn = -INFINITY;
    int bpi = 0, bni = 0;
    for (int j = 0; j < 64; ++j) {
        float dot = 0.0f;
#pragma unroll
        for (int d = 0; d < 64; ++d) dot += q[d] * rot[d * 512 + r * 64 + j];
        if (dot  > bp) { bp = dot;  bpi = j; }
        if (-dot > bn) { bn = -dot; bni = j; }
    }
    int idx = (bp >= bn) ? bpi : (64 + bni);        // first-max tie-break
    keys[(size_t)bh * FLAT_N + r * T_LEN + t] = idx + r * NBUCKETS;
}

// ------------------ stable counting sort (keys < 1024) ------------------
__global__ __launch_bounds__(256) void hist_kernel(
    const int* __restrict__ keys, int* __restrict__ blkhist, int* __restrict__ ghist)
{
    __shared__ int h[1024];
    int bh = blockIdx.y, blk = blockIdx.x, tid = threadIdx.x;
#pragma unroll
    for (int i = 0; i < 4; ++i) h[tid + 256 * i] = 0;
    __syncthreads();
    int k = keys[(size_t)bh * FLAT_N + blk * 256 + tid];
    atomicAdd(&h[k], 1);
    __syncthreads();
#pragma unroll
    for (int i = 0; i < 4; ++i) {
        int idx = tid + 256 * i;
        int cval = h[idx];
        blkhist[((size_t)(bh * NBLK + blk)) * 1024 + idx] = cval;
        if (cval) atomicAdd(&ghist[bh * 1024 + idx], cval);
    }
}

__global__ __launch_bounds__(1024) void scan_kernel(
    const int* __restrict__ ghist, int* __restrict__ goff)
{
    __shared__ int s[1024];
    int bh = blockIdx.x, tid = threadIdx.x;
    int self = ghist[bh * 1024 + tid];
    s[tid] = self;
    __syncthreads();
    for (int off = 1; off < 1024; off <<= 1) {
        int val = (tid >= off) ? s[tid - off] : 0;
        __syncthreads();
        s[tid] += val;
        __syncthreads();
    }
    goff[bh * 1024 + tid] = s[tid] - self;          // exclusive
}

__global__ __launch_bounds__(256) void blkscan_kernel(
    int* __restrict__ blkhist, const int* __restrict__ goff)
{
    int id = blockIdx.x * 256 + threadIdx.x;        // 8 * 1024 (bh,key)
    int bh = id >> 10, key = id & 1023;
    int run = goff[bh * 1024 + key];
    for (int blk = 0; blk < NBLK; ++blk) {
        size_t idx = ((size_t)(bh * NBLK + blk)) * 1024 + key;
        int cval = blkhist[idx];
        blkhist[idx] = run;
        run += cval;
    }
}

__global__ __launch_bounds__(256) void scatter_kernel(
    const int* __restrict__ keys, const int* __restrict__ blkhist,
    int* __restrict__ st_flat)
{
    __shared__ int kk[256];
    int bh = blockIdx.y, blk = blockIdx.x, tid = threadIdx.x;
    int j = blk * 256 + tid;
    int key = keys[(size_t)bh * FLAT_N + j];
    kk[tid] = key;
    __syncthreads();
    int rank = 0;
    for (int i = 0; i < tid; ++i) rank += (kk[i] == key) ? 1 : 0;
    int dest = blkhist[((size_t)(bh * NBLK + blk)) * 1024 + key] + rank;
    st_flat[(size_t)bh * FLAT_N + dest] = j;        // stable by flat index
}

// =====================================================================
// Chunked attention: one workgroup per (bh, chunk).
// dots = bq(64x64) @ bk^T(64x128) via WMMA bf16, self-mask, softmax,
// bo = P(64x128) @ bv(128x64) via WMMA, scatter per-round output + lse.
// =====================================================================
__global__ __launch_bounds__(256) void lsh_attn_kernel(
    const float* __restrict__ qk, const float* __restrict__ v,
    const int* __restrict__ st_flat,
    __hip_bfloat16* __restrict__ o_rounds, float* __restrict__ logits)
{
    __shared__ int    tq[64];
    __shared__ int    tkv[128];
    __shared__ float  nrm[128];
    __shared__ float  mrow[64];
    __shared__ float  invs[64];
    __shared__ float  Dots[64][132];
    __shared__ __bf16 Pr[64][136];

    const int c = blockIdx.x, bh = blockIdx.y, tid = threadIdx.x;
    const int lane = tid & 31, wave = tid >> 5;
    const int l = lane & 15, hi = (lane >> 4) & 1;
    const int cprev = (c + NCHUNK - 1) & (NCHUNK - 1);
    const size_t bhbase = (size_t)bh * T_LEN * 64;

    if (tid < 64) {
        int s = st_flat[(size_t)bh * FLAT_N + c * 64 + tid];
        tq[tid] = s;  tkv[tid] = s;
    } else if (tid < 128) {
        tkv[tid] = st_flat[(size_t)bh * FLAT_N + cprev * 64 + (tid - 64)];
    }
    __syncthreads();
    if (tid < 128) {                                 // key row rcp-norms
        int pos = tkv[tid] & (T_LEN - 1);
        const float* rp = qk + bhbase + (size_t)pos * 64;
        float ss = 0.0f;
#pragma unroll
        for (int d = 0; d < 64; ++d) { float xv = rp[d]; ss += xv * xv; }
        nrm[tid] = 1.0f / fmaxf(sqrtf(ss), 1e-12f);
    }
    __syncthreads();

    // ---- dots: 4x8 tiles of 16x16, K=64 (2 WMMA steps) ----
#pragma unroll
    for (int s = 0; s < 4; ++s) {
        int tt = wave * 4 + s;
        int tm = tt >> 3, tn = tt & 7;
        v8f acc;
#pragma unroll
        for (int r = 0; r < 8; ++r) acc[r] = 0.0f;
        const int qrow = tm * 16 + l;
        const int krow = tn * 16 + l;
        const float* qrp = qk + bhbase + (size_t)(tq[qrow]  & (T_LEN - 1)) * 64;
        const float* krp = qk + bhbase + (size_t)(tkv[krow] & (T_LEN - 1)) * 64;
        const float  kn  = nrm[krow];
#pragma unroll
        for (int kk = 0; kk < 64; kk += 32) {
            v16bf a, b;
#pragma unroll
            for (int j = 0; j < 8; ++j) {
                int kb = kk + hi * 16 + 2 * j;
                a[2 * j]     = (__bf16)qrp[kb];
                a[2 * j + 1] = (__bf16)qrp[kb + 1];
                int kc = kk + hi * 16 + 2 * j;
                b[2 * j]     = (__bf16)(krp[kc] * kn);
                b[2 * j + 1] = (__bf16)(krp[kc + 1] * kn);
            }
            acc = __builtin_amdgcn_wmma_f32_16x16x32_bf16(
                false, a, false, b, (short)0, acc, false, false);
        }
#pragma unroll
        for (int r = 0; r < 8; ++r)
            Dots[tm * 16 + r + hi * 8][tn * 16 + l] = acc[r];
    }
    __syncthreads();

    // ---- scale + self-mask ----
#pragma unroll
    for (int i = 0; i < 32; ++i) {
        int e = tid + 256 * i;
        int r = e >> 7, cc = e & 127;
        float dv = Dots[r][cc] * 0.125f;             // d^-0.5, d=64
        if ((tkv[cc] & (T_LEN - 1)) == (tq[r] & (T_LEN - 1))) dv = -50000.0f;
        Dots[r][cc] = dv;
    }
    __syncthreads();

    // ---- row max / sum / lse ----
    if (tid < 64) {
        float m = -INFINITY;
        for (int cc = 0; cc < 128; ++cc) m = fmaxf(m, Dots[tid][cc]);
        float sum = 0.0f;
        for (int cc = 0; cc < 128; ++cc) sum += __expf(Dots[tid][cc] - m);
        mrow[tid] = m;
        invs[tid] = 1.0f / sum;
        float lse = m + __logf(sum);
        int fl = tq[tid];
        logits[((size_t)(bh * 8 + (fl >> 13))) * T_LEN + (fl & (T_LEN - 1))] = lse;
    }
    __syncthreads();

    // ---- unnormalized probs -> bf16 ----
#pragma unroll
    for (int i = 0; i < 32; ++i) {
        int e = tid + 256 * i;
        int r = e >> 7, cc = e & 127;
        Pr[r][cc] = (__bf16)__expf(Dots[r][cc] - mrow[r]);
    }
    __syncthreads();

    // ---- bo = P @ bv : 4x4 tiles, K=128 (4 WMMA steps), scatter ----
#pragma unroll
    for (int s = 0; s < 2; ++s) {
        int tt = wave * 2 + s;
        int tm = tt >> 2, tn = tt & 3;
        v8f acc;
#pragma unroll
        for (int r = 0; r < 8; ++r) acc[r] = 0.0f;
        for (int ko = 0; ko < 128; ko += 32) {
            v16bf a, b;
#pragma unroll
            for (int j = 0; j < 8; ++j) {
                int kb = hi * 16 + 2 * j;
                a[2 * j]     = Pr[tm * 16 + l][ko + kb];
                a[2 * j + 1] = Pr[tm * 16 + l][ko + kb + 1];
                int k0i = ko + hi * 16 + 2 * j;
                int p0 = tkv[k0i]     & (T_LEN - 1);
                int p1 = tkv[k0i + 1] & (T_LEN - 1);
                b[2 * j]     = (__bf16)v[bhbase + (size_t)p0 * 64 + tn * 16 + l];
                b[2 * j + 1] = (__bf16)v[bhbase + (size_t)p1 * 64 + tn * 16 + l];
            }
            acc = __builtin_amdgcn_wmma_f32_16x16x32_bf16(
                false, a, false, b, (short)0, acc, false, false);
        }
#pragma unroll
        for (int r = 0; r < 8; ++r) {
            int m = tm * 16 + r + hi * 8;
            int fl = tq[m];
            size_t addr = (((size_t)(bh * 8 + (fl >> 13))) * T_LEN +
                           (fl & (T_LEN - 1))) * 64 + tn * 16 + l;
            o_rounds[addr] = __float2bfloat16(acc[r] * invs[m]);
        }
    }
}

// =====================================================================
// Combine rounds: softmax over 8 round-logits, weighted sum of o_rounds
// merged layout [t][head*64 + dh] feeds the output projection directly.
// =====================================================================
__global__ __launch_bounds__(256) void combine_kernel(
    const __hip_bfloat16* __restrict__ o_rounds,
    const float* __restrict__ logits, float* __restrict__ merged)
{
    int pair = blockIdx.x * 4 + (threadIdx.x >> 6);  // (bh, pos)
    int bh = pair >> 13, pos = pair & (T_LEN - 1);
    int d = threadIdx.x & 63;
    float lg[8];
    float m = -INFINITY;
#pragma unroll
    for (int r = 0; r < 8; ++r) {
        lg[r] = logits[((size_t)(bh * 8 + r)) * T_LEN + pos];
        m = fmaxf(m, lg[r]);
    }
    float s = 0.0f;
#pragma unroll
    for (int r = 0; r < 8; ++r) { lg[r] = __expf(lg[r] - m); s += lg[r]; }
    float inv = 1.0f / s;
    float acc = 0.0f;
#pragma unroll
    for (int r = 0; r < 8; ++r) {
        float ov = __bfloat162float(
            o_rounds[(((size_t)(bh * 8 + r)) * T_LEN + pos) * 64 + d]);
        acc += lg[r] * inv * ov;
    }
    merged[(size_t)pos * 512 + bh * 64 + d] = acc;
}

// ---------------- workspace layout ----------------
static constexpr size_t SZ_QK   = (size_t)NHEAD * T_LEN * 64 * 4;     // 16 MB
static constexpr size_t SZ_KEYS = (size_t)NHEAD * FLAT_N * 4;         //  2 MB
static constexpr size_t SZ_GH   = (size_t)NHEAD * 1024 * 4;           // 32 KB
static constexpr size_t SZ_BH   = (size_t)NHEAD * NBLK * 1024 * 4;    //  8 MB
static constexpr size_t SZ_LG   = (size_t)NHEAD * 8 * T_LEN * 4;      //  2 MB
static constexpr size_t SZ_OR   = (size_t)NHEAD * 8 * T_LEN * 64 * 2; // 64 MB
static constexpr size_t OFF_QK   = 0;
static constexpr size_t OFF_V    = OFF_QK + SZ_QK;
static constexpr size_t OFF_KEYS = OFF_V + SZ_QK;
static constexpr size_t OFF_ST   = OFF_KEYS + SZ_KEYS;
static constexpr size_t OFF_GH   = OFF_ST + SZ_KEYS;
static constexpr size_t OFF_GO   = OFF_GH + SZ_GH;
static constexpr size_t OFF_BH   = OFF_GO + SZ_GH;
static constexpr size_t OFF_LG   = OFF_BH + SZ_BH;
static constexpr size_t OFF_OR   = OFF_LG + SZ_LG;
static constexpr size_t OFF_MG   = OFF_OR + SZ_OR;

extern "C" void kernel_launch(void* const* d_in, const int* in_sizes, int n_in,
                              void* d_out, int out_size, void* d_ws, size_t ws_size,
                              hipStream_t stream) {
    (void)in_sizes; (void)n_in; (void)out_size; (void)ws_size;
    const float* x    = (const float*)d_in[0];
    const float* Wqk  = (const float*)d_in[1];
    const float* Wv   = (const float*)d_in[2];
    const float* Wout = (const float*)d_in[3];
    const float* rot  = (const float*)d_in[4];

    char* ws = (char*)d_ws;
    float* qkb   = (float*)(ws + OFF_QK);
    float* vb    = (float*)(ws + OFF_V);
    int*   keys  = (int*)(ws + OFF_KEYS);
    int*   stf   = (int*)(ws + OFF_ST);
    int*   ghist = (int*)(ws + OFF_GH);
    int*   goff  = (int*)(ws + OFF_GO);
    int*   blkh  = (int*)(ws + OFF_BH);
    float* lgts  = (float*)(ws + OFF_LG);
    __hip_bfloat16* orounds = (__hip_bfloat16*)(ws + OFF_OR);
    float* merged = (float*)(ws + OFF_MG);
    float* out    = (float*)d_out;

    (void)hipMemsetAsync(ghist, 0, SZ_GH, stream);

    dim3 gblk(T_LEN / 64, DIMV / 64);                // GEMM tiles 64x64
    gemm_bf16_kernel<<<gblk, 256, 0, stream>>>(x, Wqk, qkb, T_LEN, DIMV, DIMV, 1);
    gemm_bf16_kernel<<<gblk, 256, 0, stream>>>(x, Wv,  vb,  T_LEN, DIMV, DIMV, 1);

    hash_kernel<<<(NHEAD * NHASH_C * T_LEN) / 256, 256, 0, stream>>>(qkb, rot, keys);

    hist_kernel<<<dim3(NBLK, NHEAD), 256, 0, stream>>>(keys, blkh, ghist);
    scan_kernel<<<NHEAD, 1024, 0, stream>>>(ghist, goff);
    blkscan_kernel<<<(NHEAD * 1024) / 256, 256, 0, stream>>>(blkh, goff);
    scatter_kernel<<<dim3(NBLK, NHEAD), 256, 0, stream>>>(keys, blkh, stf);

    lsh_attn_kernel<<<dim3(NCHUNK, NHEAD), 256, 0, stream>>>(qkb, vb, stf, orounds, lgts);

    combine_kernel<<<(NHEAD * T_LEN) / 4, 256, 0, stream>>>(orounds, lgts, merged);

    gemm_bf16_kernel<<<gblk, 256, 0, stream>>>(merged, Wout, out, T_LEN, DIMV, DIMV, 0);
}
